// BatchGaussianRasterizer_11441792877130
// MI455X (gfx1250) — matrix operand order repaired and checked by reading, not compiled, not run on hardware
//
#include <hip/hip_runtime.h>
#include <hip/hip_bf16.h>
#include <math.h>

// ---------------- problem constants ----------------
#define H_IMG  128
#define W_IMG  128
#define HW     (H_IMG * W_IMG)     // 16384
#define PNUM   1024
#define BNUM   2
#define TANFOV 0.5f

// ---------------- WMMA vector types (gfx1250, wave32) ----------------
typedef __attribute__((ext_vector_type(16))) _Float16 v16h;
typedef __attribute__((ext_vector_type(8)))  float    v8f;

// 32-bit LDS offset of a __shared__ object (addrspace(3) ptrtoint)
#define LDS_OFFSET(p) ((unsigned)(uintptr_t)(__attribute__((address_space(3))) void*)(p))

// ---------------- workspace layout (float offsets) ----------------
// depth   : [B][P]          @ 0       (2048 floats)
// gRaw    : [B][P][12]      @ 2048    (24576 floats)
// gSorted : [B][P][12]      @ 26624   (24576 floats)   ~200KB total
#define WS_DEPTH 0
#define WS_GRAW  2048
#define WS_GSORT 26624

// per-gaussian packed attrs (12 floats):
// 0 px, 1 py, 2 conicA, 3 conicB, 4 conicC, 5 opac, 6..8 rgb, 9 origIdx(bits), 10 valid, 11 pad

// ---------------- output layout (float offsets) ----------------
#define OUT_COLOR 0         // [B][3][H][W]  (98304)
#define OUT_ALPHA 98304     // [B][1][H][W]  (32768)
#define OUT_ESTC  131072    // [B][P][3]     (6144)
#define OUT_ESTW  137216    // [B][P][1]     (2048)
#define OUT_RADII 139264    // [B][P]        (2048)

// =====================================================================
// Kernel 0: zero the atomically-accumulated est_color / est_weight region
// =====================================================================
__global__ void zero_est_kernel(float* out) {
    int gid = blockIdx.x * blockDim.x + threadIdx.x;   // 8192 threads
    if (gid < BNUM * PNUM * 4)
        out[OUT_ESTC + gid] = 0.0f;
}

// =====================================================================
// Kernel 1: per-gaussian preprocess (projection, conic, SH color, radii)
// =====================================================================
__global__ void preprocess_kernel(const float* __restrict__ means,
                                  const float* __restrict__ sh,
                                  const float* __restrict__ opac,
                                  const float* __restrict__ scales,
                                  const float* __restrict__ rots,
                                  const float* __restrict__ view,
                                  const float* __restrict__ projm,
                                  const float* __restrict__ campos,
                                  float* __restrict__ out,
                                  float* __restrict__ ws) {
    int gid = blockIdx.x * blockDim.x + threadIdx.x;
    if (gid >= BNUM * PNUM) return;
    int p = gid & (PNUM - 1);

    float mx = means[gid * 3 + 0], my = means[gid * 3 + 1], mz = means[gid * 3 + 2];

    // hom = projmatrix @ [m,1]
    float hom[4];
#pragma unroll
    for (int i = 0; i < 4; ++i)
        hom[i] = projm[i * 4 + 0] * mx + projm[i * 4 + 1] * my + projm[i * 4 + 2] * mz + projm[i * 4 + 3];
    float pw  = 1.0f / (hom[3] + 1e-7f);
    float ppx = hom[0] * pw, ppy = hom[1] * pw;

    float pv[3];
#pragma unroll
    for (int i = 0; i < 3; ++i)
        pv[i] = view[i * 4 + 0] * mx + view[i * 4 + 1] * my + view[i * 4 + 2] * mz + view[i * 4 + 3];
    float depth = pv[2];
    bool in_front = depth > 0.2f;

    // quaternion -> rotation, scale -> 3D covariance
    float qr = rots[gid * 4 + 0], qx = rots[gid * 4 + 1], qy = rots[gid * 4 + 2], qz = rots[gid * 4 + 3];
    float qn = rsqrtf(qr * qr + qx * qx + qy * qy + qz * qz);
    qr *= qn; qx *= qn; qy *= qn; qz *= qn;
    float R[3][3] = {
        {1.f - 2.f * (qy * qy + qz * qz), 2.f * (qx * qy - qr * qz),       2.f * (qx * qz + qr * qy)},
        {2.f * (qx * qy + qr * qz),       1.f - 2.f * (qx * qx + qz * qz), 2.f * (qy * qz - qr * qx)},
        {2.f * (qx * qz - qr * qy),       2.f * (qy * qz + qr * qx),       1.f - 2.f * (qx * qx + qy * qy)}};
    float s0 = scales[gid * 3 + 0], s1 = scales[gid * 3 + 1], s2 = scales[gid * 3 + 2];
    float M[3][3];
#pragma unroll
    for (int i = 0; i < 3; ++i) { M[i][0] = R[i][0] * s0; M[i][1] = R[i][1] * s1; M[i][2] = R[i][2] * s2; }
    float Sig[3][3];
#pragma unroll
    for (int i = 0; i < 3; ++i)
#pragma unroll
        for (int j = 0; j < 3; ++j)
            Sig[i][j] = M[i][0] * M[j][0] + M[i][1] * M[j][1] + M[i][2] * M[j][2];

    // EWA projection Jacobian
    const float fx = W_IMG / (2.0f * TANFOV), fy = H_IMG / (2.0f * TANFOV);
    float tz = depth, tz2 = tz * tz;
    const float lim = 1.3f * TANFOV;
    float txc = fminf(fmaxf(pv[0] / tz, -lim), lim) * tz;
    float tyc = fminf(fmaxf(pv[1] / tz, -lim), lim) * tz;
    float J[2][3] = {{fx / tz, 0.f, -fx * txc / tz2}, {0.f, fy / tz, -fy * tyc / tz2}};
    float Tm[2][3];
#pragma unroll
    for (int r = 0; r < 2; ++r)
#pragma unroll
        for (int c = 0; c < 3; ++c)
            Tm[r][c] = J[r][0] * view[0 * 4 + c] + J[r][1] * view[1 * 4 + c] + J[r][2] * view[2 * 4 + c];
    float tp[2][3];
#pragma unroll
    for (int r = 0; r < 2; ++r)
#pragma unroll
        for (int c = 0; c < 3; ++c)
            tp[r][c] = Tm[r][0] * Sig[0][c] + Tm[r][1] * Sig[1][c] + Tm[r][2] * Sig[2][c];
    float c00 = tp[0][0] * Tm[0][0] + tp[0][1] * Tm[0][1] + tp[0][2] * Tm[0][2] + 0.3f;
    float c01 = tp[0][0] * Tm[1][0] + tp[0][1] * Tm[1][1] + tp[0][2] * Tm[1][2];
    float c11 = tp[1][0] * Tm[1][0] + tp[1][1] * Tm[1][1] + tp[1][2] * Tm[1][2] + 0.3f;

    float det = c00 * c11 - c01 * c01;
    bool valid = in_front && (det > 1e-12f);
    float inv_det = 1.0f / ((det > 1e-12f) ? det : 1.0f);
    float conA = c11 * inv_det, conB = -c01 * inv_det, conC = c00 * inv_det;
    float mid = 0.5f * (c00 + c11);
    float lam = mid + sqrtf(fmaxf(mid * mid - det, 0.1f));
    float radii = valid ? ceilf(3.0f * sqrtf(lam)) : 0.0f;
    float px = ((ppx + 1.0f) * W_IMG - 1.0f) * 0.5f;
    float py = ((ppy + 1.0f) * H_IMG - 1.0f) * 0.5f;

    // SH degree-3 color eval
    float dx = mx - campos[0], dy = my - campos[1], dz = mz - campos[2];
    float dn = rsqrtf(dx * dx + dy * dy + dz * dz);
    dx *= dn; dy *= dn; dz *= dn;
    float xx = dx * dx, yy = dy * dy, zz = dz * dz;
    float xy = dx * dy, yz = dy * dz, xz = dx * dz;
    float basis[16];
    basis[0]  = 0.28209479177387814f;
    basis[1]  = -0.4886025119029199f * dy;
    basis[2]  = 0.4886025119029199f * dz;
    basis[3]  = -0.4886025119029199f * dx;
    basis[4]  = 1.0925484305920792f * xy;
    basis[5]  = -1.0925484305920792f * yz;
    basis[6]  = 0.31539156525252005f * (2.f * zz - xx - yy);
    basis[7]  = -1.0925484305920792f * xz;
    basis[8]  = 0.5462742152960396f * (xx - yy);
    basis[9]  = -0.5900435899266435f * dy * (3.f * xx - yy);
    basis[10] = 2.890611442640554f * xy * dz;
    basis[11] = -0.4570457994644658f * dy * (4.f * zz - xx - yy);
    basis[12] = 0.3731763325901154f * dz * (2.f * zz - 3.f * xx - 3.f * yy);
    basis[13] = -0.4570457994644658f * dx * (4.f * zz - xx - yy);
    basis[14] = 1.445305721320277f * dz * (xx - yy);
    basis[15] = -0.5900435899266435f * dx * (xx - 3.f * yy);
    float col[3] = {0.f, 0.f, 0.f};
#pragma unroll
    for (int k = 0; k < 16; ++k) {
        float bk = basis[k];
        col[0] += bk * sh[(gid * 16 + k) * 3 + 0];
        col[1] += bk * sh[(gid * 16 + k) * 3 + 1];
        col[2] += bk * sh[(gid * 16 + k) * 3 + 2];
    }
#pragma unroll
    for (int c = 0; c < 3; ++c) col[c] = fmaxf(col[c] + 0.5f, 0.0f);

    out[OUT_RADII + gid] = radii;
    ws[WS_DEPTH + gid]   = depth;
    float* g = ws + WS_GRAW + gid * 12;
    g[0] = px;  g[1] = py;  g[2] = conA; g[3] = conB; g[4] = conC;
    g[5] = opac[gid]; g[6] = col[0]; g[7] = col[1]; g[8] = col[2];
    g[9] = __int_as_float(p); g[10] = valid ? 1.0f : 0.0f; g[11] = 0.0f;
}

// =====================================================================
// Kernel 2: bitonic depth-sort (1024 elems, one workgroup = 32 waves)
// then gather packed attrs into depth order for coalesced staging.
// =====================================================================
__global__ __launch_bounds__(1024) void sort_kernel(float* __restrict__ ws) {
    __shared__ float key[PNUM];
    __shared__ int   val[PNUM];
    int b = blockIdx.x, tid = threadIdx.x;
    key[tid] = ws[WS_DEPTH + b * PNUM + tid];
    val[tid] = tid;
    __syncthreads();
    for (int k = 2; k <= PNUM; k <<= 1) {
        for (int j = k >> 1; j > 0; j >>= 1) {
            int ixj = tid ^ j;
            if (ixj > tid) {
                bool asc = ((tid & k) == 0);
                float ki = key[tid], kj = key[ixj];
                if ((ki > kj) == asc) {
                    key[tid] = kj; key[ixj] = ki;
                    int t = val[tid]; val[tid] = val[ixj]; val[ixj] = t;
                }
            }
            __syncthreads();
        }
    }
    int src = val[tid];
    const float* g = ws + WS_GRAW  + (b * PNUM + src) * 12;
    float*       d = ws + WS_GSORT + (b * PNUM + tid) * 12;
#pragma unroll
    for (int i = 0; i < 12; ++i) d[i] = g[i];
}

// =====================================================================
// Kernel 3: rasterize. 64 threads = 2 waves; each wave owns 32 pixels.
// Front-to-back transmittance scan over 64 chunks of 16 sorted gaussians.
// Chunk data is staged with double-buffered GLOBAL_LOAD_ASYNC_TO_LDS_B32
// (ASYNCcnt-tracked): chunk c+1 streams into LDS while chunk c computes.
// Per chunk, one v_wmma_f32_16x16x32_f16 computes
//   D[g, c] = sum_pix w[g,pix] * B[pix,c],  B cols 0..2 = target RGB, col 3 = 1
// i.e. est_color and est_weight for 16 gaussians x 32 pixels in one op.
// =====================================================================
__global__ __launch_bounds__(64) void raster_kernel(const float* __restrict__ gs,   // ws + WS_GSORT
                                                    const float* __restrict__ tgt,  // [B][3][H][W]
                                                    const float* __restrict__ bg,   // [B][3]
                                                    float* __restrict__ out) {
    __shared__ float sgbuf[2][192];       // double-buffered gaussian chunk (async-filled)
    __shared__ float wT[2][32][17];       // per-wave w transpose buffer, padded stride 17

    const int b    = blockIdx.y;
    const int tid  = threadIdx.x;
    const int wave = tid >> 5;
    const int lane = tid & 31;
    const int pixBase = blockIdx.x * 64 + wave * 32;   // 32 consecutive pixels per wave
    const int pix  = pixBase + lane;
    const float fw = (float)(pix & (W_IMG - 1));
    const float fh = (float)(pix >> 7);
    const int ncol = lane & 15;           // WMMA column (N) this lane owns
    const int lh   = lane >> 4;           // lane half (K split per ISA layout)

    const float* gsb = gs + b * PNUM * 12;
    const int NCH = PNUM / 16;

    // Per-lane flush addressing, hoisted out of the hot loop (branchless body):
    // columns 0..2 scatter into est_color (stride 3), column 3 into est_weight (stride 1).
    const bool isC       = (ncol < 3);
    const int  flushStr  = isC ? 3 : 1;
    const int  flushBase = isC ? (OUT_ESTC + b * PNUM * 3 + ncol)
                               : (OUT_ESTW + b * PNUM);

    // ---- build B matrix once: 32 pixels (K) x 16 cols (N). N: 0..2 = target RGB, 3 = ones.
    // Contiguous-K B layout: lane n = column n; lanes 0-15 hold K=0..15, lanes 16-31 K=16..31,
    // packed 2 per VGPR -> v16h element k maps to K = lh*16 + k.
    v16h bm;
#pragma unroll
    for (int k = 0; k < 16; ++k) {
        float v;
        if (ncol < 3)       v = tgt[(b * 3 + ncol) * HW + pixBase + lh * 16 + k];
        else if (ncol == 3) v = 1.0f;
        else                v = 0.0f;
        bm[k] = (_Float16)v;
    }

    // ---- issue async staging of chunk 0 into buffer 0 (3 dwords per thread)
    {
        unsigned l0 = LDS_OFFSET(&sgbuf[0][tid]);
#pragma unroll
        for (int k = 0; k < 3; ++k) {
            unsigned go = (unsigned)((tid + 64 * k) * 4);
            unsigned lo = l0 + (unsigned)(64 * k * 4);
            asm volatile("global_load_async_to_lds_b32 %0, %1, %2"
                         :: "v"(lo), "v"(go), "s"(gsb) : "memory");
        }
    }

    float T = 1.0f;
    float accR = 0.f, accG = 0.f, accB = 0.f;

    for (int chunk = 0; chunk < NCH; ++chunk) {
        const int cur = chunk & 1;
        // my async loads into sgbuf[cur] are done; barrier => everyone's are.
        asm volatile("s_wait_asynccnt 0x0" ::: "memory");
        __syncthreads();

        // kick off async staging of the next chunk into the other buffer
        if (chunk + 1 < NCH) {
            unsigned l0 = LDS_OFFSET(&sgbuf[cur ^ 1][tid]);
            unsigned gbase = (unsigned)(((chunk + 1) * 192 + tid) * 4);
#pragma unroll
            for (int k = 0; k < 3; ++k) {
                unsigned go = gbase + (unsigned)(64 * k * 4);
                unsigned lo = l0 + (unsigned)(64 * k * 4);
                asm volatile("global_load_async_to_lds_b32 %0, %1, %2"
                             :: "v"(lo), "v"(go), "s"(gsb) : "memory");
            }
        }

        const float* sg = sgbuf[cur];

        // sequential front-to-back scan over the 16 gaussians of this chunk
#pragma unroll
        for (int g = 0; g < 16; ++g) {
            const float* a = sg + g * 12;
            float ddx = a[0] - fw, ddy = a[1] - fh;
            float power = -0.5f * (a[2] * ddx * ddx + a[4] * ddy * ddy) - a[3] * ddx * ddy;
            float alpha = fminf(0.99f, a[5] * __expf(fminf(power, 0.0f)));
            bool keep = (power <= 0.0f) && (alpha >= (1.0f / 255.0f)) && (a[10] != 0.0f);
            float aa  = keep ? alpha : 0.0f;
            float wgt = aa * T;
            accR += wgt * a[6]; accG += wgt * a[7]; accB += wgt * a[8];
            T *= (1.0f - aa);
            wT[wave][lane][g] = wgt;             // pixel-major; transposed below for WMMA A
        }
        __syncthreads();

        // ---- build A matrix: 16 gaussians (M, across lanes) x 32 pixels (K).
        // A layout (matches B): lanes 0-15 row M=lane hold K=0..15 (elems 0-15);
        // lanes 16-31 row M=lane-16 hold K=16..31.
        float wv[16];
#pragma unroll
        for (int e = 0; e < 16; ++e) {
            int pk = lh * 16 + e;
            wv[e] = wT[wave][pk][ncol];
        }
        v16h am;
#pragma unroll
        for (int e = 0; e < 16; ++e) am[e] = (_Float16)wv[e];

        v8f c8 = {};
        c8 = __builtin_amdgcn_wmma_f32_16x16x32_f16(false, am, false, bm,
                                                    (short)0, c8, false, false);

        // ---- flush: D vgpr r holds M=r (lanes 0-15) / M=r+8 (lanes 16-31), N=ncol.
        // Scatter to ORIGINAL gaussian index (implements inv_order) via f32 atomics.
        // Addressing is loop-invariant per lane: idx = flushBase + origIdx * flushStr.
        if (ncol < 4) {
            int oi[8];
#pragma unroll
            for (int r = 0; r < 8; ++r)
                oi[r] = __float_as_int(sg[(r + lh * 8) * 12 + 9]);
#pragma unroll
            for (int r = 0; r < 8; ++r)
                atomicAdd(&out[flushBase + oi[r] * flushStr], c8[r]);
        }
    }

    // final composite
    out[OUT_COLOR + (b * 3 + 0) * HW + pix] = accR + T * bg[b * 3 + 0];
    out[OUT_COLOR + (b * 3 + 1) * HW + pix] = accG + T * bg[b * 3 + 1];
    out[OUT_COLOR + (b * 3 + 2) * HW + pix] = accB + T * bg[b * 3 + 2];
    out[OUT_ALPHA + b * HW + pix] = 1.0f - T;
}

// =====================================================================
extern "C" void kernel_launch(void* const* d_in, const int* in_sizes, int n_in,
                              void* d_out, int out_size, void* d_ws, size_t ws_size,
                              hipStream_t stream) {
    const float* means  = (const float*)d_in[0];
    const float* sh     = (const float*)d_in[1];
    const float* opac   = (const float*)d_in[2];
    const float* scales = (const float*)d_in[3];
    const float* rots   = (const float*)d_in[4];
    const float* tgt    = (const float*)d_in[5];
    const float* bg     = (const float*)d_in[6];
    const float* view   = (const float*)d_in[7];
    const float* projm  = (const float*)d_in[8];
    const float* campos = (const float*)d_in[9];
    float* out = (float*)d_out;
    float* ws  = (float*)d_ws;   // needs ~200KB

    zero_est_kernel<<<32, 256, 0, stream>>>(out);
    preprocess_kernel<<<(BNUM * PNUM + 255) / 256, 256, 0, stream>>>(
        means, sh, opac, scales, rots, view, projm, campos, out, ws);
    sort_kernel<<<BNUM, 1024, 0, stream>>>(ws);
    dim3 rgrid(HW / 64, BNUM);
    raster_kernel<<<rgrid, 64, 0, stream>>>(ws + WS_GSORT, tgt, bg, out);
}
